// PDPLoss_38749194945146
// MI455X (gfx1250) — matrix operand order repaired and checked by reading, not compiled, not run on hardware
//
#include <hip/hip_runtime.h>
#include <hip/hip_bf16.h>

// ---------------------------------------------------------------------------
// PDP loss: IFFT-as-GEMM on WMMA (gfx1250).
//   pdp[r,k] = |(1/2046) * sum_n x[r,n] e^{+2pi i k n / 2046}|^2 , k=0..1023
//   (half spectrum; real input => mirror symmetry folds all reductions)
// Single fused pass: A block = 16 pred rows + 16 tgt rows (R=2 row tiles),
// every B fragment loaded once feeds 2 WMMAs -> 0.5 KB fresh B per WMMA
// (halves L2 traffic vs the two-pass version).  K loop kept rolled.
// ---------------------------------------------------------------------------

typedef __attribute__((ext_vector_type(16))) _Float16 v16h;
typedef __attribute__((ext_vector_type(8)))  _Float16 v8h;
typedef __attribute__((ext_vector_type(8)))  float    v8f;

#define FFT_N        2046
#define K_IN         1024
#define NHALF        1024
#define ROWS         8192          // 32*64*4
#define KTILES       32            // 1024 / 32
#define NTILES       64            // 1024 / 16
#define WAVES        16
#define NT_PER_WAVE  4             // 64 ntiles / 16 waves
#define WG_ROWS      16
#define NBLOCKS      (ROWS / WG_ROWS)   // 512
#define FRAG_HALVES  512           // 32x16 f16 fragment = 512 halves = 1 KiB

// workspace byte offsets
#define WS_BCOS_BYTES (KTILES * NTILES * FRAG_HALVES * 2)   // 2 MiB
#define WS_MSE_OFF    (2 * WS_BCOS_BYTES)                   // 4 MiB
#define WS_DLY_OFF    (WS_MSE_OFF + NBLOCKS * 4)

// ---- deterministic wave32 tree reductions (fixed mask order) --------------
__device__ __forceinline__ float halfsum16(float v) {   // sum within 16-lane group
#pragma unroll
  for (int m = 8; m; m >>= 1) v += __shfl_xor(v, m, 32);
  return v;
}
__device__ __forceinline__ float halfmax16(float v) {
#pragma unroll
  for (int m = 8; m; m >>= 1) v = fmaxf(v, __shfl_xor(v, m, 32));
  return v;
}
__device__ __forceinline__ float halfmin16(float v) {
#pragma unroll
  for (int m = 8; m; m >>= 1) v = fminf(v, __shfl_xor(v, m, 32));
  return v;
}
__device__ __forceinline__ float wavesum32(float v) {
#pragma unroll
  for (int m = 16; m; m >>= 1) v += __shfl_xor(v, m, 32);
  return v;
}

// ---------------------------------------------------------------------------
// Twiddle generation: cos/sin matrices directly in WMMA B-fragment layout.
// Fragment (kt,nt): lane l, half h -> K = kt*32 + (l<16?0:16) + h,
//                                     N = nt*16 + (l&15).
// Angle exact via integer (k*n) % 2046.
// ---------------------------------------------------------------------------
__global__ void gen_twiddles(_Float16* __restrict__ bcos,
                             _Float16* __restrict__ bsin) {
  int gid  = blockIdx.x * blockDim.x + threadIdx.x;   // 65536 threads
  int lane = gid & 31;
  int frag = gid >> 5;                                // 0..2047
  int kt   = frag >> 6;
  int nt   = frag & 63;
  int n    = nt * 16 + (lane & 15);
  int kb   = kt * 32 + ((lane & 16) ? 16 : 0);
  v16h c, s;
#pragma unroll
  for (int h = 0; h < 16; ++h) {
    int k = kb + h;
    int r = (k * n) % FFT_N;                          // exact in int
    float ang = 6.283185307179586f * ((float)r * (1.0f / (float)FFT_N));
    float sv, cv;
    __sincosf(ang, &sv, &cv);
    c[h] = (_Float16)cv;
    s[h] = (_Float16)sv;
  }
  int off = frag * FRAG_HALVES + lane * 16;           // 32B per lane
  *(v16h*)(bcos + off) = c;
  *(v16h*)(bsin + off) = s;
}

// ---------------------------------------------------------------------------
// Main fused kernel: 512 threads (16 waves), 16 row indices per WG, pred and
// tgt processed together.  Wave w owns bins n in [w*64, w*64+64).
// acc[0..3]=cos/pred acc[4..7]=sin/pred acc[8..11]=cos/tgt acc[12..15]=sin/tgt
// Logical row lr = tensor*16 + m  (0..31).
// ---------------------------------------------------------------------------
__launch_bounds__(512)
__global__ void pdp_loss_main(const float* __restrict__ pred,
                              const float* __restrict__ tgt,
                              const _Float16* __restrict__ bcos,
                              const _Float16* __restrict__ bsin,
                              float* __restrict__ mse_wg,
                              float* __restrict__ delay_row) {
  __shared__ _Float16 ldsA[2 * WG_ROWS * K_IN];   // 64 KiB (pred | tgt)
  __shared__ float    scratch[32 * WAVES];        // 2 KiB per-row wave partials
  __shared__ float    msep[WAVES];
  __shared__ int      red_max[32], red_min[32];
  __shared__ float    red_p0[32], red_p1023[32];
  __shared__ float    red_e0[32], red_e1023[32];
  __shared__ float    sc_row[32], sm_scale[32], sm_off[32];
  __shared__ float    dly_all[32];

  const int tid  = threadIdx.x;
  const int lane = tid & 31;
  const int wave = tid >> 5;
  const int hw   = (lane & 16) ? 1 : 0;   // half-wave select
  const int l15  = lane & 15;
  const int row0 = blockIdx.x * WG_ROWS;

  if (tid < 32) {
    red_max[tid] = 0;                     // pdp >= 0 -> int-order works
    red_min[tid] = 0x7f7fffff;            // +FLT_MAX bits
    red_p0[tid] = 0.f;  red_p1023[tid] = 0.f;
    red_e0[tid] = 0.f;  red_e1023[tid] = 0.f;
  }
  // ---- stage A: 16 contiguous rows of each tensor, f32 -> f16 in LDS ----
  {
    const float4* sp4 = (const float4*)(pred + (size_t)row0 * K_IN);
    const float4* st4 = (const float4*)(tgt  + (size_t)row0 * K_IN);
    uint2* dst = (uint2*)ldsA;
    for (int i = tid; i < WG_ROWS * K_IN / 4; i += 512) {
      float4 v = sp4[i];
      union { _Float16 h[4]; uint2 u; } p;
      p.h[0] = (_Float16)v.x; p.h[1] = (_Float16)v.y;
      p.h[2] = (_Float16)v.z; p.h[3] = (_Float16)v.w;
      dst[i] = p.u;
      float4 w = st4[i];
      p.h[0] = (_Float16)w.x; p.h[1] = (_Float16)w.y;
      p.h[2] = (_Float16)w.z; p.h[3] = (_Float16)w.w;
      dst[WG_ROWS * K_IN / 4 + i] = p.u;
    }
  }
  __syncthreads();

  // ---- WMMA GEMM: K loop over 32 k-tiles (rolled; avoid spills) ----
  v8f acc[16] = {};
  const _Float16* bc  = bcos + (wave * NT_PER_WAVE) * FRAG_HALVES + lane * 16;
  const _Float16* bs  = bsin + (wave * NT_PER_WAVE) * FRAG_HALVES + lane * 16;
  const _Float16* apP = ldsA + l15 * K_IN + (hw ? 16 : 0);
  const _Float16* apT = apP + WG_ROWS * K_IN;
#pragma unroll 1
  for (int kt = 0; kt < KTILES; ++kt) {
    // A fragment (16x32 f16) contiguous:
    //   lanes<16: halves0-15=K0-15; lanes>=16: K16-31
    v8h alo = *(const v8h*)(apP);
    v8h ahi = *(const v8h*)(apP + 8);
    v8h tlo = *(const v8h*)(apT);
    v8h thi = *(const v8h*)(apT + 8);
    apP += 32; apT += 32;
    v16h aP, aT;
#pragma unroll
    for (int i = 0; i < 8; ++i) {
      aP[i] = alo[i]; aP[i + 8] = ahi[i];
      aT[i] = tlo[i]; aT[i + 8] = thi[i];
    }
#pragma unroll
    for (int t = 0; t < NT_PER_WAVE; ++t) {
      v16h bcf = *(const v16h*)(bc + t * FRAG_HALVES);
      acc[t] = __builtin_amdgcn_wmma_f32_16x16x32_f16(
          false, aP, false, bcf, (short)0, acc[t], false, false);
      acc[8 + t] = __builtin_amdgcn_wmma_f32_16x16x32_f16(
          false, aT, false, bcf, (short)0, acc[8 + t], false, false);
      v16h bsf = *(const v16h*)(bs + t * FRAG_HALVES);
      acc[4 + t] = __builtin_amdgcn_wmma_f32_16x16x32_f16(
          false, aP, false, bsf, (short)0, acc[4 + t], false, false);
      acc[12 + t] = __builtin_amdgcn_wmma_f32_16x16x32_f16(
          false, aT, false, bsf, (short)0, acc[12 + t], false, false);
    }
    bc += NTILES * FRAG_HALVES;           // next k-tile row of fragments
    bs += NTILES * FRAG_HALVES;
  }

  // ---- PDP (C layout: row m = j + hw*8, col n = wave*64 + t*16 + l15) ----
  const float invN2 = 1.0f / ((float)FFT_N * (float)FFT_N);
  float pdp[2][NT_PER_WAVE][8];
#pragma unroll
  for (int tt = 0; tt < 2; ++tt)
#pragma unroll
    for (int t = 0; t < NT_PER_WAVE; ++t)
#pragma unroll
      for (int j = 0; j < 8; ++j) {
        float re = acc[tt * 8 + t][j], im = acc[tt * 8 + 4 + t][j];
        pdp[tt][t][j] = (re * re + im * im) * invN2;
      }

  // ---- per-row energy/max/min via shuffle trees ----
#pragma unroll
  for (int tt = 0; tt < 2; ++tt) {
#pragma unroll
    for (int j = 0; j < 8; ++j) {
      int m  = j + hw * 8;
      int lr = tt * 16 + m;
      float e = 0.f, mx = 0.f, mn = 3.4e38f;
#pragma unroll
      for (int t = 0; t < NT_PER_WAVE; ++t) {
        float v = pdp[tt][t][j];
        e += v; mx = fmaxf(mx, v); mn = fminf(mn, v);
      }
      e  = halfsum16(e);
      mx = halfmax16(mx);
      mn = halfmin16(mn);
      if (l15 == 0) {
        scratch[lr * WAVES + wave] = e;
        atomicMax(&red_max[lr], __float_as_int(mx));  // order-independent
        atomicMin(&red_min[lr], __float_as_int(mn));
      }
    }
    if (wave == 0 && l15 == 0) {          // n == 0
#pragma unroll
      for (int j = 0; j < 8; ++j) red_p0[tt * 16 + j + hw * 8] = pdp[tt][0][j];
    }
    if (wave == WAVES - 1 && l15 == 15) { // n == 1023
#pragma unroll
      for (int j = 0; j < 8; ++j)
        red_p1023[tt * 16 + j + hw * 8] = pdp[tt][NT_PER_WAVE - 1][j];
    }
  }
  __syncthreads();

  // ---- per-row normalization + softmax params (fixed-order sum) ----
  if (tid < 32) {
    float Sh = 0.f;
    for (int i = 0; i < WAVES; ++i) Sh += scratch[tid * WAVES + i];
    float E = 2.f * Sh - red_p0[tid] - red_p1023[tid];  // full 2046-bin energy
    float scale = (E < 1e-8f) ? 1.f : 1.f / E;
    float mxn = __int_as_float(red_max[tid]) * scale;
    float mnn = __int_as_float(red_min[tid]) * scale;
    float rng = fmaxf(mxn - mnn, 1e-8f);
    float s10 = 10.f / rng;                             // TEMP_SCALE / rng
    sc_row[tid]   = scale;
    sm_scale[tid] = s10 * scale;                        // applies to raw pdp
    sm_off[tid]   = s10 * mxn;                          // stability shift
  }
  __syncthreads();

  // ---- exp phase: softmax partial sums for both tensors ----
#pragma unroll
  for (int tt = 0; tt < 2; ++tt) {
#pragma unroll
    for (int j = 0; j < 8; ++j) {
      int m  = j + hw * 8;
      int lr = tt * 16 + m;
      float ss = sm_scale[lr], so = sm_off[lr];
      float z = 0.f;
#pragma unroll
      for (int t = 0; t < NT_PER_WAVE; ++t) {
        int n = wave * 64 + t * 16 + l15;
        float e = __expf(fmaf(pdp[tt][t][j], ss, -so));
        z += e;
        if (n == 0)         red_e0[lr] = e;
        if (n == NHALF - 1) red_e1023[lr] = e;
      }
      z = halfsum16(z);
      if (l15 == 0) scratch[lr * WAVES + wave] = z;
    }
  }
  // ---- MSE elementwise (both normalized pdps live in registers) ----
  {
    float macc = 0.f;
#pragma unroll
    for (int j = 0; j < 8; ++j) {
      int m = j + hw * 8;
      float scP = sc_row[m], scT = sc_row[16 + m];
#pragma unroll
      for (int t = 0; t < NT_PER_WAVE; ++t) {
        int n = wave * 64 + t * 16 + l15;
        float d = pdp[0][t][j] * scP - pdp[1][t][j] * scT;
        float w = (n == 0 || n == NHALF - 1) ? 1.f : 2.f; // mirror weight
        macc = fmaf(w, d * d, macc);
      }
    }
    macc = wavesum32(macc);
    if (lane == 0) msep[wave] = macc;
  }
  __syncthreads();

  // ---- soft-delay via folded sums; per-row delay diff; WG MSE ----
  if (tid < 32) {
    float Zh = 0.f;
    for (int i = 0; i < WAVES; ++i) Zh += scratch[tid * WAVES + i];
    float e0 = red_e0[tid], e1 = red_e1023[tid];
    float Zf  = 2.f * Zh - e0 - e1;
    float num = (float)FFT_N * (Zh - e0) - 1023.f * e1;
    dly_all[tid] = num / Zf;
  }
  __syncthreads();
  if (tid < WG_ROWS)
    delay_row[row0 + tid] = fabsf(dly_all[tid] - dly_all[16 + tid]);
  if (tid == 0) {
    float s = 0.f;
    for (int i = 0; i < WAVES; ++i) s += msep[i];
    mse_wg[blockIdx.x] = s;
  }
}

// ---------------------------------------------------------------------------
// Deterministic final reduction.  mse_loss = S_mse/(8192*2046);
// delay_loss = S_dly/(8192*2046) (|diff|/FFT_N folded in) -> same denom.
// ---------------------------------------------------------------------------
__global__ void finalize(const float* __restrict__ mse_wg,
                         const float* __restrict__ delay_row,
                         float* __restrict__ out) {
  __shared__ float red[256];
  int tid = threadIdx.x;
  float s_mse = 0.f;
  for (int i = tid; i < NBLOCKS; i += 256) s_mse += mse_wg[i];
  float s_dly = 0.f;
  for (int i = tid; i < ROWS; i += 256) s_dly += delay_row[i];
  red[tid] = 0.7f * s_mse + 0.3f * s_dly;
  __syncthreads();
  for (int s = 128; s > 0; s >>= 1) {
    if (tid < s) red[tid] += red[tid + s];
    __syncthreads();
  }
  if (tid == 0) out[0] = red[0] * (1.0f / (8192.0f * 2046.0f));
}

// ---------------------------------------------------------------------------
extern "C" void kernel_launch(void* const* d_in, const int* in_sizes, int n_in,
                              void* d_out, int out_size, void* d_ws, size_t ws_size,
                              hipStream_t stream) {
  (void)in_sizes; (void)n_in; (void)out_size; (void)ws_size;
  const float* pred = (const float*)d_in[0];
  const float* tgt  = (const float*)d_in[1];

  _Float16* bcos = (_Float16*)d_ws;
  _Float16* bsin = bcos + KTILES * NTILES * FRAG_HALVES;
  float* mse_wg    = (float*)((char*)d_ws + WS_MSE_OFF);
  float* delay_row = (float*)((char*)d_ws + WS_DLY_OFF);

  gen_twiddles<<<(KTILES * NTILES * 32) / 256, 256, 0, stream>>>(bcos, bsin);
  pdp_loss_main<<<NBLOCKS, 512, 0, stream>>>(pred, tgt, bcos, bsin,
                                             mse_wg, delay_row);
  finalize<<<1, 256, 0, stream>>>(mse_wg, delay_row, (float*)d_out);
}
